// MultiHeadAttention_11751030522181
// MI455X (gfx1250) — matrix operand order repaired and checked by reading, not compiled, not run on hardware
//
#include <hip/hip_runtime.h>
#include <hip/hip_bf16.h>

// Problem constants (match reference)
#define BATCH 2
#define SEQ   2048
#define DMODEL 1024
#define NHEADS 16
#define DKDIM 64
#define NTOK (BATCH * SEQ)          // 4096
#define BH   (BATCH * NHEADS)       // 32

typedef __attribute__((ext_vector_type(16))) __bf16 v16bf;
typedef __attribute__((ext_vector_type(8)))  __bf16 v8bf;
typedef __attribute__((ext_vector_type(8)))  float  v8f;
typedef __attribute__((ext_vector_type(4)))  float  f32x4;

static __device__ __forceinline__ v8f wmma_bf16(v16bf a, v16bf b, v8f c) {
  // D = A(16x32 bf16) * B(32x16 bf16) + C(16x16 f32)
  return __builtin_amdgcn_wmma_f32_16x16x32_bf16(false, a, false, b,
                                                 (short)0, c, false, false);
}

// ---------------------------------------------------------------------------
// Repack a [1024 x 1024] f32 row-major weight (W[k][c], Y = X @ W) into bf16
// B-fragment order: slot g = (kt*64 + ct)*32 + lane holds 16 contiguous bf16:
//   lane<16 : K = kt*32 + j (j<8), kt*32 + 16 + (j-8) (j>=8), col = ct*16 + (lane&15)
//   lane>=16: K = kt*32 + 8 + j (j<8), kt*32 + 24 + (j-8) (j>=8), same col
// ---------------------------------------------------------------------------
__global__ __launch_bounds__(256) void wprep_kernel(const float* __restrict__ W,
                                                    __bf16* __restrict__ Wf) {
  const int g    = blockIdx.x * 256 + threadIdx.x;   // 0 .. 65535
  const int lane = g & 31;
  const int ct   = (g >> 5) & 63;
  const int kt   = g >> 11;                          // 0 .. 31
  const int kb   = (lane & 16) ? 8 : 0;
  const int c    = ct * 16 + (lane & 15);
  __bf16* dst = Wf + (size_t)g * 16;
#pragma unroll
  for (int j = 0; j < 16; ++j)
    dst[j] = (__bf16)W[(size_t)(kt * 32 + kb + (j < 8 ? j : j + 8)) * DMODEL + c];
}

// ---------------------------------------------------------------------------
// Projection GEMM: Y[r,c] = sum_k X[r,k]*W[k,c] + b[c]
//   mode 0/1: store bf16 to [b,h,s,64] (Q, K)
//   mode 2  : store bf16 transposed to [b,h,64,s] (V^T for the AV GEMM)
// One 16x16 tile per wave, K-loop of 32 steps.
// ---------------------------------------------------------------------------
__global__ __launch_bounds__(256) void proj_kernel(const float* __restrict__ X,
                                                   const __bf16* __restrict__ Wf,
                                                   const float* __restrict__ bias,
                                                   __bf16* __restrict__ out,
                                                   int mode) {
  const int wid  = threadIdx.x >> 5;
  const int lane = threadIdx.x & 31;
  const int tile = blockIdx.x * 8 + wid;   // 0 .. 16383
  const int rt   = tile >> 6;
  const int ct   = tile & 63;
  const int r0   = rt << 4;
  const int c0   = ct << 4;
  const int l15  = lane & 15;
  const int hi   = (lane >> 4) & 1;
  const int kb1  = hi ? 8 : 0;

  const float* arow = X + (size_t)(r0 + l15) * DMODEL;
  v8f acc = {};
  for (int kt = 0; kt < 32; ++kt) {
    const int k = kt << 5;
    const float* ap = arow + k + kb1;
    f32x4 a0 = *(const f32x4*)(ap);
    f32x4 a1 = *(const f32x4*)(ap + 4);
    f32x4 a2 = *(const f32x4*)(ap + 16);
    f32x4 a3 = *(const f32x4*)(ap + 20);
    v16bf a;
#pragma unroll
    for (int j = 0; j < 4; ++j) {
      a[j]      = (__bf16)a0[j];
      a[4 + j]  = (__bf16)a1[j];
      a[8 + j]  = (__bf16)a2[j];
      a[12 + j] = (__bf16)a3[j];
    }
    v16bf b = *(const v16bf*)(Wf + (((size_t)kt * 64 + ct) * 32 + lane) * 16);
    acc = wmma_bf16(a, b, acc);
  }

  const int   col  = c0 + l15;
  const float bv   = bias[col];
  const int   roff = hi ? 8 : 0;
  const int   h    = col >> 6;
  const int   d    = col & 63;

  if (mode == 2) {
    // V^T: [b,h,64,s]; rows for this lane are 8 contiguous s values.
    const int token0 = r0 + roff;
    const int bidx   = token0 >> 11;
    const int s0     = token0 & (SEQ - 1);
    __bf16* dst = out + (((size_t)(bidx * NHEADS + h) * DKDIM + d) * SEQ + s0);
    v8bf vv;
#pragma unroll
    for (int v = 0; v < 8; ++v) vv[v] = (__bf16)(acc[v] + bv);
    *(v8bf*)dst = vv;
  } else {
#pragma unroll
    for (int v = 0; v < 8; ++v) {
      const int token = r0 + roff + v;
      const int bidx  = token >> 11;
      const int s     = token & (SEQ - 1);
      out[((size_t)(bidx * NHEADS + h) * SEQ + s) * DKDIM + d] =
          (__bf16)(acc[v] + bv);
    }
  }
}

// ---------------------------------------------------------------------------
// Fused attention: one block (8 waves) owns (bh, 16 query rows).
//   LDS: P panel 16 x 2048 f32 (128 KB) + O accum 16 x 64 f32 + rowinv[16].
//   Phase 1: each wave WMMAs its 16x256 score chunk into LDS.
//   Phase 2: per-row softmax in LDS; single normalized f32 write to d_out.
//   Phase 3: each wave WMMAs P_exp(LDS, scaled by rowinv on cvt) x V^T over
//            its k-range; partials reduced with LDS atomicAdd; bf16 O out.
// The normalized-weights store is the only large HBM write; K/V stay in L2.
// ---------------------------------------------------------------------------
__global__ __launch_bounds__(256) void attn_kernel(const __bf16* __restrict__ Q,
                                                   const __bf16* __restrict__ K,
                                                   const __bf16* __restrict__ Vt,
                                                   float* __restrict__ att,
                                                   __bf16* __restrict__ O) {
  extern __shared__ float smem[];
  float* Pl     = smem;                       // 16 x 2048
  float* Oacc   = smem + 16 * SEQ;            // 16 x 64
  float* rowinv = Oacc + 16 * DKDIM;          // 16

  const int tid  = threadIdx.x;
  const int wid  = tid >> 5;
  const int lane = tid & 31;
  const int bh   = blockIdx.x >> 7;           // 0..31
  const int qt   = blockIdx.x & 127;          // 0..127
  const int q0   = qt << 4;
  const int l15  = lane & 15;
  const int hi   = (lane >> 4) & 1;
  const int kb1  = hi ? 16 : 0;
  const int kb   = hi ? 16 : 0;
  const int roff = hi ? 8 : 0;

  // Zero the O accumulator (covered by the phase-1 barrier).
  for (int i = tid; i < 16 * DKDIM; i += 256) Oacc[i] = 0.f;

  // ---- Phase 1: scores into LDS -------------------------------------------
  const __bf16* qrow = Q + ((size_t)bh * SEQ + q0 + l15) * DKDIM;
  v16bf aq0, aq1;  // Q A-fragments for d=0..31 and d=32..63 (loaded once)
  {
    v8bf t0 = *(const v8bf*)(qrow + kb1);
    v8bf t1 = *(const v8bf*)(qrow + kb1 + 8);
    v8bf t2 = *(const v8bf*)(qrow + 32 + kb1);
    v8bf t3 = *(const v8bf*)(qrow + 32 + kb1 + 8);
#pragma unroll
    for (int j = 0; j < 8; ++j) {
      aq0[j] = t0[j]; aq0[8 + j] = t1[j];
      aq1[j] = t2[j]; aq1[8 + j] = t3[j];
    }
  }
  const float scale = 0.125f;  // 1/sqrt(64)
  for (int t = 0; t < 16; ++t) {
    const int k0 = (wid * 16 + t) << 4;       // this wave's 16 k-tiles
    const __bf16* krow = K + ((size_t)bh * SEQ + k0 + l15) * DKDIM;
    v16bf b0 = *(const v16bf*)(krow + kb);
    v16bf b1 = *(const v16bf*)(krow + 32 + kb);
    v8f acc = {};
    acc = wmma_bf16(aq0, b0, acc);
    acc = wmma_bf16(aq1, b1, acc);
    float* pl = Pl + (size_t)roff * SEQ + k0 + l15;
#pragma unroll
    for (int v = 0; v < 8; ++v) pl[(size_t)v * SEQ] = acc[v] * scale;
  }
  __syncthreads();

  // ---- Phase 2: softmax (wave w owns rows 2w, 2w+1) -----------------------
  {
    const int r = 2 * wid + hi;
    float* prow = Pl + (size_t)r * SEQ;
    float m = -3.402823466e+38f;
    for (int j = l15; j < SEQ; j += 16) m = fmaxf(m, prow[j]);
#pragma unroll
    for (int mk = 8; mk > 0; mk >>= 1) m = fmaxf(m, __shfl_xor(m, mk, 32));
    float s = 0.f;
    for (int j = l15; j < SEQ; j += 16) {
      const float e = __expf(prow[j] - m);
      prow[j] = e;                 // keep exp() in LDS for phase 3
      s += e;
    }
#pragma unroll
    for (int mk = 8; mk > 0; mk >>= 1) s += __shfl_xor(s, mk, 32);
    const float inv = 1.0f / s;
    if (l15 == 0) rowinv[r] = inv;
    // Single normalized f32 write of this panel row to global output.
    float* grow = att + ((size_t)bh * SEQ + q0 + r) * SEQ;
    for (int j = l15; j < SEQ; j += 16) grow[j] = prow[j] * inv;
  }
  __syncthreads();

  // ---- Phase 3: O partial = (P_exp * rowinv) x V^T over this wave's k-range
  const float ainv = rowinv[l15];  // A-fragment row is fixed per lane
#pragma unroll
  for (int dt = 0; dt < 4; ++dt) {
    const int d0 = dt << 4;
    const __bf16* vrow = Vt + ((size_t)bh * DKDIM + d0 + l15) * SEQ;
    v8f acc = {};
#pragma unroll
    for (int t = 0; t < 8; ++t) {
      const int k = (wid * 8 + t) << 5;       // wid*256 + t*32
      const float* ap = Pl + (size_t)l15 * SEQ + k + kb1;
      f32x4 x0 = *(const f32x4*)(ap);
      f32x4 x1 = *(const f32x4*)(ap + 4);
      f32x4 x2 = *(const f32x4*)(ap + 8);
      f32x4 x3 = *(const f32x4*)(ap + 12);
      v16bf a;
#pragma unroll
      for (int j = 0; j < 4; ++j) {
        a[j]      = (__bf16)(x0[j] * ainv);
        a[4 + j]  = (__bf16)(x1[j] * ainv);
        a[8 + j]  = (__bf16)(x2[j] * ainv);
        a[12 + j] = (__bf16)(x3[j] * ainv);
      }
      v16bf b = *(const v16bf*)(vrow + k + kb);
      acc = wmma_bf16(a, b, acc);
    }
#pragma unroll
    for (int v = 0; v < 8; ++v)
      atomicAdd(&Oacc[(roff + v) * DKDIM + d0 + l15], acc[v]);
  }
  __syncthreads();

  // ---- Store O tile (16 x 64) as bf16 row-major [token, DMODEL] -----------
  const int bidx = bh >> 4;
  const int h    = bh & 15;
  for (int i = tid; i < 16 * DKDIM; i += 256) {
    const int rr = i >> 6;
    const int cc = i & 63;
    const int token = bidx * SEQ + q0 + rr;
    O[(size_t)token * DMODEL + h * DKDIM + cc] = (__bf16)Oacc[i];
  }
}

// ---------------------------------------------------------------------------
// Final GEMM: out[r,c] = sum_k O[r,k]*Wo[k,c] + bo[c], f32 result to d_out.
// ---------------------------------------------------------------------------
__global__ __launch_bounds__(256) void out_kernel(const __bf16* __restrict__ O,
                                                  const __bf16* __restrict__ Wf,
                                                  const float* __restrict__ bias,
                                                  float* __restrict__ out) {
  const int wid  = threadIdx.x >> 5;
  const int lane = threadIdx.x & 31;
  const int tile = blockIdx.x * 8 + wid;   // 0 .. 16383
  const int rt   = tile >> 6;
  const int ct   = tile & 63;
  const int r0   = rt << 4;
  const int c0   = ct << 4;
  const int l15  = lane & 15;
  const int hi   = (lane >> 4) & 1;
  const int kb1  = hi ? 8 : 0;

  const __bf16* arow = O + (size_t)(r0 + l15) * DMODEL;
  v8f acc = {};
  for (int kt = 0; kt < 32; ++kt) {
    const int k = kt << 5;
    v8bf a0 = *(const v8bf*)(arow + k + kb1);
    v8bf a1 = *(const v8bf*)(arow + k + kb1 + 16);
    v16bf a;
#pragma unroll
    for (int j = 0; j < 8; ++j) { a[j] = a0[j]; a[8 + j] = a1[j]; }
    v16bf b = *(const v16bf*)(Wf + (((size_t)kt * 64 + ct) * 32 + lane) * 16);
    acc = wmma_bf16(a, b, acc);
  }

  const int   col  = c0 + l15;
  const float bv   = bias[col];
  const int   roff = hi ? 8 : 0;
#pragma unroll
  for (int v = 0; v < 8; ++v)
    out[(size_t)(r0 + roff + v) * DMODEL + col] = acc[v] + bv;
}

// ---------------------------------------------------------------------------
extern "C" void kernel_launch(void* const* d_in, const int* in_sizes, int n_in,
                              void* d_out, int out_size, void* d_ws, size_t ws_size,
                              hipStream_t stream) {
  (void)in_sizes; (void)n_in; (void)out_size; (void)ws_size;
  const float* q  = (const float*)d_in[0];
  const float* k  = (const float*)d_in[1];
  const float* v  = (const float*)d_in[2];
  const float* Wq = (const float*)d_in[3];
  const float* bq = (const float*)d_in[4];
  const float* Wk = (const float*)d_in[5];
  const float* bk = (const float*)d_in[6];
  const float* Wv = (const float*)d_in[7];
  const float* bv = (const float*)d_in[8];
  const float* Wo = (const float*)d_in[9];
  const float* bo = (const float*)d_in[10];

  // Workspace (bf16 elems): 4 weight frags (1M each) + Q,K,Vt,O (4M each) = 40 MB
  __bf16* Wqf = (__bf16*)d_ws;
  __bf16* Wkf = Wqf + (1u << 20);
  __bf16* Wvf = Wkf + (1u << 20);
  __bf16* Wof = Wvf + (1u << 20);
  __bf16* Qb  = Wof + (1u << 20);
  __bf16* Kb  = Qb  + (4u << 20);
  __bf16* Vtb = Kb  + (4u << 20);
  __bf16* Ob  = Vtb + (4u << 20);

  float* out = (float*)d_out;                  // [2,2048,1024]
  float* att = out + (size_t)NTOK * DMODEL;    // [2,16,2048,2048]

  // 1) Repack weights into bf16 B-fragment order
  wprep_kernel<<<256, 256, 0, stream>>>(Wq, Wqf);
  wprep_kernel<<<256, 256, 0, stream>>>(Wk, Wkf);
  wprep_kernel<<<256, 256, 0, stream>>>(Wv, Wvf);
  wprep_kernel<<<256, 256, 0, stream>>>(Wo, Wof);

  // 2) Q/K/V projections
  proj_kernel<<<2048, 256, 0, stream>>>(q, Wqf, bq, Qb, 0);
  proj_kernel<<<2048, 256, 0, stream>>>(k, Wkf, bk, Kb, 1);
  proj_kernel<<<2048, 256, 0, stream>>>(v, Wvf, bv, Vtb, 2);

  // 3) Fused scores + softmax + P*V. One block per (bh, 16-query stripe).
  //    Dynamic LDS: 16x2048 P panel + 16x64 O accum + rowinv[16].
  const size_t shmem = (size_t)(16 * SEQ + 16 * DKDIM + 16) * sizeof(float);
  attn_kernel<<<BH * (SEQ / 16), 256, shmem, stream>>>(Qb, Kb, Vtb, att, Ob);

  // 4) out = O Wo + bo  (f32 result)
  out_kernel<<<2048, 256, 0, stream>>>(Ob, Wof, bo, out);
}
